// TemporalGaitGCN_28097676051061
// MI455X (gfx1250) — matrix operand order repaired and checked by reading, not compiled, not run on hardware
//
#include <hip/hip_runtime.h>
#include <hip/hip_bf16.h>
#include <cmath>

// ---------------------------------------------------------------------------
// TemporalGaitGCN forward for gfx1250 (MI455X).
// All dense GEMMs run on v_wmma_f32_16x16x32_f16 (f16 in, f32 accumulate).
// Weights are converted+transposed to f16 (N x K) once per launch so every
// WMMA B-fragment load is two contiguous 8-half runs per lane (wave32 layout).
// Two GEMM engines:
//   gemm16  : 1 wave / 32x32 tile, fragments direct from global (attention
//             shapes: N=800 or K=64/800).
//   gemm_tdm: 2 waves / 32x64 tile; BOTH the 32-row A slab and the 64-row Bt
//             slab are staged into LDS by the Tensor Data Mover
//             (tensor_load_to_lds, double buffered, s_wait_tensorcnt); the
//             compute loop is pure ds_load_b128 -> v_wmma with zero global
//             traffic. Used for the dominant K%128==0 projection/FFN GEMMs.
// ---------------------------------------------------------------------------

typedef _Float16     v16h __attribute__((ext_vector_type(16)));
typedef _Float16     v8h  __attribute__((ext_vector_type(8)));
typedef float        v8f  __attribute__((ext_vector_type(8)));
typedef unsigned int v4u  __attribute__((ext_vector_type(4)));
typedef int          v8i  __attribute__((ext_vector_type(8)));
typedef int          v4i  __attribute__((ext_vector_type(4)));

#if defined(__has_builtin)
#  if __has_builtin(__builtin_amdgcn_tensor_load_to_lds)
#    define HAVE_TDM 1
#  endif
#endif
#ifndef HAVE_TDM
#  define HAVE_TDM 0
#endif

#if HAVE_TDM
#  pragma message("CDNA5: tensor_load_to_lds TDM path ENABLED")
#else
#  pragma message("CDNA5: TDM builtin unavailable -- cooperative-copy fallback")
#endif

#define B_  8
#define T_  32
#define J_  25
#define N_  800
#define FD_ 64
#define GO_ 128
#define KSLAB 128

// Wave32 fragment layouts per CDNA5 ISA 7.12.2 (16-bit, 16x16x32):
//   A (16x32 f16): lane L holds row M=L%16; kh=L/16; regs 0..3 -> K =
//     kh*8 + 0..7, regs 4..7 -> K = 16 + kh*8 + 0..7  (two 16B loads)
//   B (32x16 f16): lane L holds col N=L%16; same K order as A (two 16B loads)
//   C/D (16x16 f32): reg r: M = kh*8 + r, N = lane%16
__device__ __forceinline__ v16h load_a_frag(const _Float16* __restrict__ row,
                                            int kk, int kh)
{
    v8h lo = *(const v8h*)(row + kk + kh * 8);
    v8h hi = *(const v8h*)(row + kk + 16 + kh * 8);
    v16h r;
#pragma unroll
    for (int i = 0; i < 8; ++i) { r[i] = lo[i]; r[i + 8] = hi[i]; }
    return r;
}

__device__ __forceinline__ v16h load_b_frag(const _Float16* __restrict__ row,
                                            int kk, int kh)
{
    v8h lo = *(const v8h*)(row + kk + kh * 8);
    v8h hi = *(const v8h*)(row + kk + 16 + kh * 8);
    v16h r;
#pragma unroll
    for (int i = 0; i < 8; ++i) { r[i] = lo[i]; r[i + 8] = hi[i]; }
    return r;
}

#if HAVE_TDM
// Issue one TDM 2-D tile load: rows x KSLAB halfs, row stride = stride_elems,
// destination LDS byte address lds_addr. D# per CDNA5 ISA 8.3/8.4:
//   group0: count=1 | lds_addr | global_addr(57b) | type=2
//   group1: data_size=1 (2B) | tensor dims = tile dims | dim0 stride
__device__ __forceinline__ void tdm_load_2d(const void* gptr, unsigned lds_addr,
                                            int rows, int stride_elems)
{
    const unsigned long long ga = (unsigned long long)(uintptr_t)gptr;
    v4u g0 = { 1u, lds_addr, (unsigned)ga,
               (unsigned)((ga >> 32) & 0x01FFFFFFu) | 0x80000000u };
    v8i g1 = { (int)0x00010000,                 // data_size=1 (2 bytes)
               (int)((KSLAB & 0xFFFF) << 16),   // tensor_dim0 lo
               (int)(rows << 16),               // dim0 hi | tensor_dim1 lo
               (int)(KSLAB << 16),              // dim1 hi | tile_dim0
               rows,                            // tile_dim1 | tile_dim2=0
               stride_elems,                    // tensor_dim0_stride[31:0]
               0, 0 };
    v4i z4 = { 0, 0, 0, 0 };
#if defined(__clang_major__) && (__clang_major__ >= 23)
    v8i z8 = { 0, 0, 0, 0, 0, 0, 0, 0 };
    __builtin_amdgcn_tensor_load_to_lds(g0, g1, z4, z4, z8, 0);
#else
    __builtin_amdgcn_tensor_load_to_lds(g0, g1, z4, z4, 0);
#endif
}
#endif

// Shared epilogue: write one wave's 32x32 tile.
// flags: bit0 = relu, bit1 = accumulate into existing C32
__device__ __forceinline__ void
store_tile(v8f acc[2][2], int tM, int tN, int l16, int kh,
           const float* __restrict__ bias,
           float* __restrict__ C32, _Float16* __restrict__ C16,
           _Float16* __restrict__ C16T,
           int ldc, int M, int flags, float scale)
{
#pragma unroll
    for (int sn = 0; sn < 2; ++sn) {
        const int col = tN * 32 + sn * 16 + l16;
        const float bval = bias ? bias[col] : 0.0f;
#pragma unroll
        for (int sm = 0; sm < 2; ++sm) {
#pragma unroll
            for (int r = 0; r < 8; ++r) {
                const int m = tM * 32 + sm * 16 + kh * 8 + r;
                float v = acc[sm][sn][r] * scale + bval;
                if (flags & 2) v += C32[(size_t)m * ldc + col];
                if (flags & 1) v = fmaxf(v, 0.0f);
                if (C32)  C32[(size_t)m * ldc + col] = v;
                if (C16)  C16[(size_t)m * ldc + col] = (_Float16)v;
                if (C16T) C16T[(size_t)col * M + m]  = (_Float16)v; // transposed
            }
        }
    }
}

// ---------------------------------------------------------------------------
// gemm16: C(M,N) = scale * A(MxK) * Bt(NxK)^T [+ bias] [+ C] [relu]
// One wave -> 32x32 output tile (2x2 accumulators), direct global fragments.
// M, N multiples of 32; K multiple of 32; lda/ldb multiples of 8 halfs.
// ---------------------------------------------------------------------------
__global__ void __launch_bounds__(32)
gemm16(const _Float16* __restrict__ A, int lda,
       const _Float16* __restrict__ Bt, int ldb,
       const float* __restrict__ bias,
       float* __restrict__ C32,
       _Float16* __restrict__ C16,
       _Float16* __restrict__ C16T,
       int ldc, int M, int N, int K, int flags, float scale)
{
    const int tN = blockIdx.x, tM = blockIdx.y;
    const int lane = threadIdx.x;
    const int l16 = lane & 15, kh = lane >> 4;

    const _Float16* arow0 = A  + (size_t)(tM * 32 + l16) * lda;
    const _Float16* arow1 = arow0 + (size_t)16 * lda;
    const _Float16* brow0 = Bt + (size_t)(tN * 32 + l16) * ldb;
    const _Float16* brow1 = brow0 + (size_t)16 * ldb;

    v8f acc[2][2] = {};
    for (int kk = 0; kk < K; kk += 32) {
        v16h a0 = load_a_frag(arow0, kk, kh);
        v16h a1 = load_a_frag(arow1, kk, kh);
        v16h b0 = load_b_frag(brow0, kk, kh);
        v16h b1 = load_b_frag(brow1, kk, kh);
        acc[0][0] = __builtin_amdgcn_wmma_f32_16x16x32_f16(
                        false, a0, false, b0, (short)0, acc[0][0], false, false);
        acc[0][1] = __builtin_amdgcn_wmma_f32_16x16x32_f16(
                        false, a0, false, b1, (short)0, acc[0][1], false, false);
        acc[1][0] = __builtin_amdgcn_wmma_f32_16x16x32_f16(
                        false, a1, false, b0, (short)0, acc[1][0], false, false);
        acc[1][1] = __builtin_amdgcn_wmma_f32_16x16x32_f16(
                        false, a1, false, b1, (short)0, acc[1][1], false, false);
        if (kk + 64 < K) {                 // global_prefetch_b8 next slab
            __builtin_prefetch(arow0 + kk + 64, 0, 1);
            __builtin_prefetch(arow1 + kk + 64, 0, 1);
            __builtin_prefetch(brow0 + kk + 64, 0, 1);
            __builtin_prefetch(brow1 + kk + 64, 0, 1);
        }
    }
    store_tile(acc, tM, tN, l16, kh, bias, C32, C16, C16T, ldc, M, flags, scale);
}

// ---------------------------------------------------------------------------
// gemm_tdm: same math; block = 2 waves, 32 rows x 64 cols. A slab (32xKSLAB)
// and Bt slab (64xKSLAB) both staged into LDS by TDM, double buffered; the
// compute loop issues only ds_load_b128 + v_wmma.
// Requires K % KSLAB == 0, N % 64 == 0, M % 32 == 0.
// ---------------------------------------------------------------------------
__global__ void __launch_bounds__(64)
gemm_tdm(const _Float16* __restrict__ A, int lda,
         const _Float16* __restrict__ Bt, int ldb,
         const float* __restrict__ bias,
         float* __restrict__ C32,
         _Float16* __restrict__ C16,
         _Float16* __restrict__ C16T,
         int ldc, int M, int N, int K, int flags, float scale)
{
    __shared__ _Float16 As[2][32 * KSLAB];     // 2 x  8 KB
    __shared__ _Float16 Bs[2][64 * KSLAB];     // 2 x 16 KB

    const int wave = threadIdx.x >> 5;
    const int lane = threadIdx.x & 31;
    const int l16 = lane & 15, kh = lane >> 4;
    const int tM = blockIdx.y;
    const int tN = blockIdx.x * 2 + wave;
    const int nslab = K / KSLAB;

    // Stage slab s of A (32 rows) and Bt (64 rows) into buffer buf.
    auto stage = [&](int s, int buf) {
        const _Float16* asrc = A  + (size_t)(tM * 32) * lda + (size_t)s * KSLAB;
        const _Float16* bsrc = Bt + (size_t)(blockIdx.x * 64) * ldb + (size_t)s * KSLAB;
#if HAVE_TDM
        if (threadIdx.x < 32) {
            // Flat LDS addresses truncate to the LDS byte offset (ISA 10.2).
            tdm_load_2d(asrc, (unsigned)(uintptr_t)&As[buf][0], 32, lda);
            tdm_load_2d(bsrc, (unsigned)(uintptr_t)&Bs[buf][0], 64, ldb);
        }
#else
        for (int idx = threadIdx.x; idx < 32 * KSLAB / 8; idx += 64) {
            int row = (idx * 8) / KSLAB, cb = (idx * 8) % KSLAB;
            *(v8h*)&As[buf][row * KSLAB + cb] =
                *(const v8h*)(asrc + (size_t)row * lda + cb);
        }
        for (int idx = threadIdx.x; idx < 64 * KSLAB / 8; idx += 64) {
            int row = (idx * 8) / KSLAB, cb = (idx * 8) % KSLAB;
            *(v8h*)&Bs[buf][row * KSLAB + cb] =
                *(const v8h*)(bsrc + (size_t)row * ldb + cb);
        }
#endif
    };
    auto stage_wait = [&]() {
#if HAVE_TDM
        if (threadIdx.x < 32) __builtin_amdgcn_s_wait_tensorcnt(0);
#endif
        __syncthreads();
    };

    v8f acc[2][2] = {};
    stage(0, 0);
    for (int s = 0; s < nslab; ++s) {
        const int buf = s & 1;
        stage_wait();                               // slab s resident
        if (s + 1 < nslab) stage(s + 1, buf ^ 1);   // DMA overlaps compute

        const _Float16* a0row = &As[buf][l16 * KSLAB];
        const _Float16* a1row = a0row + 16 * KSLAB;
        const _Float16* b0row = &Bs[buf][(wave * 32 + l16) * KSLAB];
        const _Float16* b1row = b0row + 16 * KSLAB;
#pragma unroll
        for (int kks = 0; kks < KSLAB; kks += 32) {
            v16h a0 = load_a_frag(a0row, kks, kh);  // ds_load_b128
            v16h a1 = load_a_frag(a1row, kks, kh);
            v16h b0 = load_b_frag(b0row, kks, kh);
            v16h b1 = load_b_frag(b1row, kks, kh);
            acc[0][0] = __builtin_amdgcn_wmma_f32_16x16x32_f16(
                            false, a0, false, b0, (short)0, acc[0][0], false, false);
            acc[0][1] = __builtin_amdgcn_wmma_f32_16x16x32_f16(
                            false, a0, false, b1, (short)0, acc[0][1], false, false);
            acc[1][0] = __builtin_amdgcn_wmma_f32_16x16x32_f16(
                            false, a1, false, b0, (short)0, acc[1][0], false, false);
            acc[1][1] = __builtin_amdgcn_wmma_f32_16x16x32_f16(
                            false, a1, false, b1, (short)0, acc[1][1], false, false);
        }
        __syncthreads();                            // reads done before reuse
    }
    store_tile(acc, tM, tN, l16, kh, bias, C32, C16, C16T, ldc, M, flags, scale);
}

// f32 (K x N) weight -> f16 transposed (N x K)
__global__ void cvt_wt(const float* __restrict__ W, _Float16* __restrict__ Wt,
                       int K, int N)
{
    int tid = blockIdx.x * blockDim.x + threadIdx.x;
    if (tid >= K * N) return;
    int n = tid / K, k = tid % K;
    Wt[tid] = (_Float16)W[(size_t)k * N + n];
}

// Feature extractor: h = relu(relu(x@w1+b1)@w2+b2), one block per row.
__global__ void __launch_bounds__(64)
fe_kernel(const float* __restrict__ x,
          const float* __restrict__ w1, const float* __restrict__ b1,
          const float* __restrict__ w2, const float* __restrict__ b2,
          _Float16* __restrict__ h16)
{
    const int row = blockIdx.x, c = threadIdx.x;
    __shared__ float xs[7];
    __shared__ float h1[64];
    if (c < 7) xs[c] = x[row * 7 + c];
    __syncthreads();
    float a = b1[c];
#pragma unroll
    for (int k = 0; k < 7; ++k) a += xs[k] * w1[k * 64 + c];
    h1[c] = fmaxf(a, 0.0f);
    __syncthreads();
    float a2 = b2[c];
#pragma unroll
    for (int k = 0; k < 64; ++k) a2 += h1[k] * w2[k * 64 + c];
    h16[row * 64 + c] = (_Float16)fmaxf(a2, 0.0f);
}

// Gumbel mask: mask = same && (logit + g1 > g0), scores pre-scaled by 1/64.
__global__ void mask_kernel(const float* __restrict__ scores,
                            const float* __restrict__ gu,
                            float* __restrict__ maskf)
{
    int idx = blockIdx.x * blockDim.x + threadIdx.x;
    if (idx >= N_ * N_) return;
    int i = idx / N_, j = idx % N_;
    float u0 = gu[(size_t)idx * 2 + 0];
    float u1 = gu[(size_t)idx * 2 + 1];
    float g0 = -__logf(-__logf(u0));
    float g1 = -__logf(-__logf(u1));
    bool same = (i / J_) == (j / J_);
    float logit = same ? scores[idx] : -1.0e9f;
    maskf[idx] = (same && (logit + g1 > g0)) ? 1.0f : 0.0f;
}

// Masked row softmax: alpha = softmax(where(mask, att, -1e9)) * mask -> f16
__global__ void __launch_bounds__(256)
msoftmax(const float* __restrict__ att, const float* __restrict__ maskf,
         _Float16* __restrict__ alpha16)
{
    const int i = blockIdx.x, t = threadIdx.x;
    __shared__ float red[256];
    const float* arow = att + (size_t)i * N_;
    const float* mrow = maskf + (size_t)i * N_;
    float mx = -3.4e38f;
    for (int j = t; j < N_; j += 256) {
        float v = (mrow[j] != 0.0f) ? arow[j] : -1.0e9f;
        mx = fmaxf(mx, v);
    }
    red[t] = mx; __syncthreads();
    for (int s = 128; s > 0; s >>= 1) { if (t < s) red[t] = fmaxf(red[t], red[t + s]); __syncthreads(); }
    mx = red[0]; __syncthreads();
    float sum = 0.0f;
    for (int j = t; j < N_; j += 256) {
        float v = (mrow[j] != 0.0f) ? arow[j] : -1.0e9f;
        sum += __expf(v - mx);
    }
    red[t] = sum; __syncthreads();
    for (int s = 128; s > 0; s >>= 1) { if (t < s) red[t] += red[t + s]; __syncthreads(); }
    const float inv = 1.0f / red[0];
    for (int j = t; j < N_; j += 256) {
        float v = (mrow[j] != 0.0f) ? arow[j] : -1.0e9f;
        alpha16[(size_t)i * N_ + j] = (_Float16)(__expf(v - mx) * inv * mrow[j]);
    }
}

// BatchNorm stats over rows (axis 0): one block per column.
__global__ void __launch_bounds__(256)
bnstats(const float* __restrict__ X, int M, int D,
        float* __restrict__ mean, float* __restrict__ var)
{
    const int col = blockIdx.x, t = threadIdx.x;
    __shared__ float s1[256], s2[256];
    float a = 0.0f, b = 0.0f;
    for (int r = t; r < M; r += 256) {
        float v = X[(size_t)r * D + col];
        a += v; b += v * v;
    }
    s1[t] = a; s2[t] = b; __syncthreads();
    for (int s = 128; s > 0; s >>= 1) {
        if (t < s) { s1[t] += s1[t + s]; s2[t] += s2[t + s]; }
        __syncthreads();
    }
    if (t == 0) {
        float mu = s1[0] / M;
        mean[col] = mu;
        var[col] = s2[0] / M - mu * mu;
    }
}

// y = relu((x-mu)*rsqrt(var+eps)*g + b) [+ resid]; emit f32 and/or f16.
__global__ void bnapply(const float* __restrict__ X,
                        const float* __restrict__ mean, const float* __restrict__ var,
                        const float* __restrict__ g, const float* __restrict__ b,
                        const float* __restrict__ resid,
                        float* __restrict__ out32, _Float16* __restrict__ out16,
                        int M, int D)
{
    int idx = blockIdx.x * blockDim.x + threadIdx.x;
    if (idx >= M * D) return;
    int col = idx % D;
    float v = (X[idx] - mean[col]) * rsqrtf(var[col] + 1e-5f) * g[col] + b[col];
    v = fmaxf(v, 0.0f);
    if (resid) v += resid[idx];
    if (out32) out32[idx] = v;
    out16[idx] = (_Float16)v;
}

// gate scalar head: val = y1(row) . w2 + b2
__global__ void __launch_bounds__(64)
gate2_kernel(const float* __restrict__ y1, const float* __restrict__ w2,
             const float* __restrict__ b2, float* __restrict__ gateval)
{
    const int row = blockIdx.x, t = threadIdx.x;
    __shared__ float red[64];
    red[t] = y1[row * 64 + t] * w2[t];
    __syncthreads();
    for (int s = 32; s > 0; s >>= 1) { if (t < s) red[t] += red[t + s]; __syncthreads(); }
    if (t == 0) gateval[row] = red[0] + b2[0];
}

// frames[t,d] = sum_j softmax_j(gate[t,:]) * x10[t*J+j, d]
__global__ void __launch_bounds__(128)
pool_kernel(const float* __restrict__ gateval, const float* __restrict__ x10,
            float* __restrict__ framesOut)
{
    const int t = blockIdx.x, d = threadIdx.x;
    __shared__ float a[J_];
    if (d < J_) a[d] = gateval[t * J_ + d];
    __syncthreads();
    if (d == 0) {
        float mx = -3.4e38f;
        for (int j = 0; j < J_; ++j) mx = fmaxf(mx, a[j]);
        float s = 0.0f;
        for (int j = 0; j < J_; ++j) { a[j] = __expf(a[j] - mx); s += a[j]; }
        for (int j = 0; j < J_; ++j) a[j] /= s;
    }
    __syncthreads();
    float acc = 0.0f;
    for (int j = 0; j < J_; ++j)
        acc += a[j] * x10[(size_t)(t * J_ + j) * GO_ + d];
    framesOut[t * GO_ + d] = acc;
}

// z = frames + positional encoding; emit f32 + f16.
__global__ void zinit(const float* __restrict__ frames,
                      float* __restrict__ z32, _Float16* __restrict__ z16)
{
    int idx = blockIdx.x * blockDim.x + threadIdx.x;
    if (idx >= B_ * T_ * GO_) return;
    int d = idx % GO_;
    int t = (idx / GO_) % T_;
    int pair = d >> 1;
    float div = __expf((float)(2 * pair) * (-__logf(10000.0f) / (float)GO_));
    float ang = (float)t * div;
    float pe = (d & 1) ? __cosf(ang) : __sinf(ang);
    float v = frames[idx] + pe;
    z32[idx] = v; z16[idx] = (_Float16)v;
}

// Tiny 4-head encoder self-attention over T=32: one wave per (batch, head).
__global__ void __launch_bounds__(32)
enc_att(const float* __restrict__ qkv, _Float16* __restrict__ o16)
{
    const int bh = blockIdx.x;
    const int b = bh >> 2, h = bh & 3;
    const int t = threadIdx.x;
    const float* base = qkv + (size_t)(b * T_) * 384;
    const float* qrow = base + t * 384 + h * 32;
    float att[32];
    float mx = -3.4e38f;
#pragma unroll 4
    for (int s = 0; s < 32; ++s) {
        const float* krow = base + s * 384 + 128 + h * 32;
        float acc = 0.0f;
        for (int d = 0; d < 32; ++d) acc += qrow[d] * krow[d];
        acc *= 0.17677669529663687f;   // 1/sqrt(32)
        att[s] = acc; mx = fmaxf(mx, acc);
    }
    float sum = 0.0f;
    for (int s = 0; s < 32; ++s) { att[s] = __expf(att[s] - mx); sum += att[s]; }
    const float inv = 1.0f / sum;
    for (int d = 0; d < 32; ++d) {
        float acc = 0.0f;
        for (int s = 0; s < 32; ++s) acc += att[s] * base[s * 384 + 256 + h * 32 + d];
        o16[(size_t)(b * T_ + t) * GO_ + h * 32 + d] = (_Float16)(acc * inv);
    }
}

// LayerNorm over last dim (128); one block per row; updates z32 and z16.
__global__ void __launch_bounds__(128)
ln_kernel(float* __restrict__ z32, _Float16* __restrict__ z16,
          const float* __restrict__ g, const float* __restrict__ b)
{
    const int row = blockIdx.x, d = threadIdx.x;
    __shared__ float red[128];
    float v = z32[row * GO_ + d];
    red[d] = v; __syncthreads();
    for (int s = 64; s > 0; s >>= 1) { if (d < s) red[d] += red[d + s]; __syncthreads(); }
    float mu = red[0] / (float)GO_; __syncthreads();
    float c = v - mu;
    red[d] = c * c; __syncthreads();
    for (int s = 64; s > 0; s >>= 1) { if (d < s) red[d] += red[d + s]; __syncthreads(); }
    float var = red[0] / (float)GO_;
    float y = c * rsqrtf(var + 1e-5f) * g[d] + b[d];
    z32[row * GO_ + d] = y; z16[row * GO_ + d] = (_Float16)y;
}

// Final head: out = relu(z_last @ fc_w1 + b1) @ fc_w2 + b2; one block per batch.
__global__ void __launch_bounds__(64)
fc_kernel(const float* __restrict__ z32,
          const float* __restrict__ w1, const float* __restrict__ b1,
          const float* __restrict__ w2, const float* __restrict__ b2,
          float* __restrict__ out)
{
    const int b = blockIdx.x, c = threadIdx.x;
    const float* row = z32 + (size_t)(b * T_ + (T_ - 1)) * GO_;
    __shared__ float h[64];
    float acc = b1[c];
    for (int k = 0; k < GO_; ++k) acc += row[k] * w1[k * 64 + c];
    h[c] = fmaxf(acc, 0.0f);
    __syncthreads();
    if (c < 2) {
        float o = b2[c];
        for (int k = 0; k < 64; ++k) o += h[k] * w2[k * 2 + c];
        out[b * 2 + c] = o;
    }
}

// ---------------------------------------------------------------------------
extern "C" void kernel_launch(void* const* d_in, const int* in_sizes, int n_in,
                              void* d_out, int out_size, void* d_ws, size_t ws_size,
                              hipStream_t stream)
{
    (void)in_sizes; (void)n_in; (void)out_size; (void)ws_size;

    const float* x  = (const float*)d_in[0];
    const float* gu = (const float*)d_in[1];
    auto P = [&](int i) { return (const float*)d_in[i]; };

    // ---- workspace carve-out -------------------------------------------
    char* ws = (char*)d_ws;
    size_t off = 0;
    auto allocF = [&](size_t n) { void* p = ws + off; off = (off + n * 4 + 255) & ~(size_t)255; return (float*)p; };
    auto allocH = [&](size_t n) { void* p = ws + off; off = (off + n * 2 + 255) & ~(size_t)255; return (_Float16*)p; };

    _Float16* wt_qw = allocH(512 * 64);
    _Float16* wt_kw = allocH(512 * 64);
    _Float16* wt_g1[4]; for (int i = 0; i < 4; ++i) wt_g1[i] = allocH(1024 * 64);
    _Float16* wt_g2[4]; for (int i = 0; i < 4; ++i) wt_g2[i] = allocH(1024 * 1024);
    _Float16* wt_g3[4]; for (int i = 0; i < 4; ++i) wt_g3[i] = allocH(1024 * 1024);
    _Float16* wt_g10[4]; for (int i = 0; i < 4; ++i) wt_g10[i] = allocH(128 * 1024);
    _Float16* wt_gate1 = allocH(64 * 128);
    _Float16* wt_in[2], *wt_out[2], *wt_ff1[2], *wt_ff2[2];
    for (int l = 0; l < 2; ++l) {
        wt_in[l]  = allocH(384 * 128);
        wt_out[l] = allocH(128 * 128);
        wt_ff1[l] = allocH(128 * 128);
        wt_ff2[l] = allocH(128 * 128);
    }

    _Float16* h16   = allocH(N_ * 64);
    _Float16* q16   = allocH(N_ * 512);
    _Float16* k16   = allocH(N_ * 512);
    float*    scores = allocF((size_t)N_ * N_);
    float*    maskf  = allocF((size_t)N_ * N_);
    _Float16* tq16  = allocH((size_t)N_ * 1024);
    _Float16* tk16  = allocH((size_t)N_ * 1024);
    _Float16* vT16  = allocH((size_t)1024 * N_);
    float*    tco   = allocF((size_t)N_ * 1024);
    float*    att   = allocF((size_t)N_ * N_);
    _Float16* alpha16 = allocH((size_t)N_ * N_);
    float*    bn_mean = allocF(1024);
    float*    bn_var  = allocF(1024);
    float*    x1_32 = allocF((size_t)N_ * 1024);
    _Float16* x1_16 = allocH((size_t)N_ * 1024);
    _Float16* x2_16 = allocH((size_t)N_ * 1024);
    _Float16* x3_16 = allocH((size_t)N_ * 1024);
    float*    x10_32 = allocF(N_ * GO_);
    _Float16* x10_16 = allocH(N_ * GO_);
    float*    gate1  = allocF(N_ * 64);
    float*    gateval = allocF(N_);
    float*    frames  = allocF(B_ * T_ * GO_);
    float*    z32 = allocF(B_ * T_ * GO_);
    _Float16* z16 = allocH(B_ * T_ * GO_);
    float*    qkv = allocF(B_ * T_ * 384);
    _Float16* o16 = allocH(B_ * T_ * GO_);
    _Float16* ff16 = allocH(B_ * T_ * GO_);

    // ---- launch helpers ------------------------------------------------
    auto gemm = [&](const _Float16* A, int lda, const _Float16* Bt, int ldb,
                    const float* bias, float* C32, _Float16* C16, _Float16* C16T,
                    int ldc, int M, int Nn, int K, int flags, float scale) {
        if ((K % KSLAB == 0) && (Nn % 64 == 0)) {
            gemm_tdm<<<dim3(Nn / 64, M / 32), dim3(64), 0, stream>>>(
                A, lda, Bt, ldb, bias, C32, C16, C16T, ldc, M, Nn, K, flags, scale);
        } else {
            gemm16<<<dim3(Nn / 32, M / 32), dim3(32), 0, stream>>>(
                A, lda, Bt, ldb, bias, C32, C16, C16T, ldc, M, Nn, K, flags, scale);
        }
    };
    auto cvt = [&](int idx, int K, int Nn, _Float16* dst) {
        int total = K * Nn;
        cvt_wt<<<dim3((total + 255) / 256), dim3(256), 0, stream>>>(P(idx), dst, K, Nn);
    };

    // ---- weight conversion (f32 KxN -> f16 NxK) ------------------------
    cvt(6, 64, 512, wt_qw);  cvt(8, 64, 512, wt_kw);
    for (int i = 0; i < 4; ++i) cvt(10 + 2 * i, 64, 1024, wt_g1[i]);
    for (int i = 0; i < 4; ++i) cvt(18 + 2 * i, 1024, 1024, wt_g2[i]);
    for (int i = 0; i < 4; ++i) cvt(26 + 2 * i, 1024, 1024, wt_g3[i]);
    for (int i = 0; i < 4; ++i) cvt(34 + 2 * i, 1024, 128, wt_g10[i]);
    cvt(50, 128, 64, wt_gate1);
    for (int l = 0; l < 2; ++l) {
        int base = 54 + l * 12;
        cvt(base + 0, 128, 384, wt_in[l]);
        cvt(base + 2, 128, 128, wt_out[l]);
        cvt(base + 4, 128, 128, wt_ff1[l]);
        cvt(base + 6, 128, 128, wt_ff2[l]);
    }

    // masked multi-head graph attention block
    auto tconv = [&](const _Float16* Xin, int Din, int Hh, int Dout,
                     _Float16* const* wqkvs,
                     const float* bq, const float* bk, const float* bv, const float* bs) {
        const int Dtot = Hh * Dout;
        gemm(Xin, Din, wqkvs[0], Din, bq, nullptr, tq16, nullptr, Dtot, N_, Dtot, Din, 0, 1.0f);
        gemm(Xin, Din, wqkvs[1], Din, bk, nullptr, tk16, nullptr, Dtot, N_, Dtot, Din, 0, 1.0f);
        gemm(Xin, Din, wqkvs[2], Din, bv, nullptr, nullptr, vT16, Dtot, N_, Dtot, Din, 0, 1.0f);
        gemm(Xin, Din, wqkvs[3], Din, bs, tco, nullptr, nullptr, Dtot, N_, Dtot, Din, 0, 1.0f);
        const float iscale = 1.0f / sqrtf((float)Dout);
        for (int h = 0; h < Hh; ++h) {
            gemm(tq16 + h * Dout, Dtot, tk16 + h * Dout, Dtot, nullptr,
                 att, nullptr, nullptr, N_, N_, N_, Dout, 0, iscale);
            msoftmax<<<dim3(N_), dim3(256), 0, stream>>>(att, maskf, alpha16);
            gemm(alpha16, N_, vT16 + (size_t)h * Dout * N_, N_, nullptr,
                 tco + h * Dout, nullptr, nullptr, Dtot, N_, Dout, N_, 2, 1.0f);
        }
    };
    auto bnblock = [&](int D, const float* g, const float* bb, const float* resid,
                       float* out32, _Float16* out16) {
        bnstats<<<dim3(D), dim3(256), 0, stream>>>(tco, N_, D, bn_mean, bn_var);
        int total = N_ * D;
        bnapply<<<dim3((total + 255) / 256), dim3(256), 0, stream>>>(
            tco, bn_mean, bn_var, g, bb, resid, out32, out16, N_, D);
    };

    // ---- per-batch pipeline -------------------------------------------
    for (int b = 0; b < B_; ++b) {
        fe_kernel<<<dim3(N_), dim3(64), 0, stream>>>(
            x + (size_t)b * N_ * 7, P(2), P(3), P(4), P(5), h16);

        gemm(h16, 64, wt_qw, 64, P(7), nullptr, q16, nullptr, 512, N_, 512, 64, 0, 1.0f);
        gemm(h16, 64, wt_kw, 64, P(9), nullptr, k16, nullptr, 512, N_, 512, 64, 0, 1.0f);
        gemm(q16, 512, k16, 512, nullptr, scores, nullptr, nullptr, N_, N_, N_, 512, 0, 1.0f / 64.0f);
        mask_kernel<<<dim3((N_ * N_ + 255) / 256), dim3(256), 0, stream>>>(
            scores, gu + (size_t)b * N_ * N_ * 2, maskf);

        tconv(h16, 64, 8, 128, wt_g1, P(11), P(13), P(15), P(17));
        bnblock(1024, P(42), P(43), nullptr, x1_32, x1_16);              // x1
        tconv(x1_16, 1024, 8, 128, wt_g2, P(19), P(21), P(23), P(25));
        bnblock(1024, P(44), P(45), x1_32, nullptr, x2_16);              // x2 = relu(bn)+x1
        tconv(x2_16, 1024, 8, 128, wt_g3, P(27), P(29), P(31), P(33));
        bnblock(1024, P(46), P(47), nullptr, nullptr, x3_16);            // x3
        tconv(x3_16, 1024, 1, 128, wt_g10, P(35), P(37), P(39), P(41));
        bnblock(128, P(48), P(49), nullptr, x10_32, x10_16);             // x10

        gemm(x10_16, 128, wt_gate1, 128, P(51), gate1, nullptr, nullptr, 64, N_, 64, 128, 1, 1.0f);
        gate2_kernel<<<dim3(N_), dim3(64), 0, stream>>>(gate1, P(52), P(53), gateval);
        pool_kernel<<<dim3(T_), dim3(128), 0, stream>>>(
            gateval, x10_32, frames + (size_t)b * T_ * GO_);
    }

    // ---- transformer encoder over frames ------------------------------
    zinit<<<dim3((B_ * T_ * GO_ + 255) / 256), dim3(256), 0, stream>>>(frames, z32, z16);
    for (int l = 0; l < 2; ++l) {
        int base = 54 + l * 12;
        gemm(z16, 128, wt_in[l], 128, P(base + 1), qkv, nullptr, nullptr, 384, B_ * T_, 384, 128, 0, 1.0f);
        enc_att<<<dim3(B_ * 4), dim3(32), 0, stream>>>(qkv, o16);
        gemm(o16, 128, wt_out[l], 128, P(base + 3), z32, nullptr, nullptr, 128, B_ * T_, 128, 128, 2, 1.0f);
        ln_kernel<<<dim3(B_ * T_), dim3(128), 0, stream>>>(z32, z16, P(base + 8), P(base + 9));
        gemm(z16, 128, wt_ff1[l], 128, P(base + 5), nullptr, ff16, nullptr, 128, B_ * T_, 128, 128, 1, 1.0f);
        gemm(ff16, 128, wt_ff2[l], 128, P(base + 7), z32, nullptr, nullptr, 128, B_ * T_, 128, 128, 2, 1.0f);
        ln_kernel<<<dim3(B_ * T_), dim3(128), 0, stream>>>(z32, z16, P(base + 10), P(base + 11));
    }

    fc_kernel<<<dim3(B_), dim3(64), 0, stream>>>(z32, P(78), P(79), P(80), P(81), (float*)d_out);
}
